// RNN_16647293240165
// MI455X (gfx1250) — matrix operand order repaired and checked by reading, not compiled, not run on hardware
//
#include <hip/hip_runtime.h>
#include <hip/hip_bf16.h>

#define SEQ 512
#define BSZ 128
#define DIN 256
#define DH  512

#define WI_LDK (DIN + 8)   // pad 8 bf16 to break LDS bank conflicts
#define WH_LDK (DH + 8)

typedef __attribute__((ext_vector_type(16))) __bf16 v16bf;
typedef __attribute__((ext_vector_type(8)))  __bf16 v8bf;
typedef __attribute__((ext_vector_type(8)))  float  v8f;

// ---------------------------------------------------------------------------
// CDNA5 helpers
// ---------------------------------------------------------------------------
__device__ inline unsigned lds_off_u32(const void* p) {
    // Flat LDS addresses map to LDS by truncation (see 00_overview)
    return (unsigned)(unsigned long long)p;
}

__device__ inline void async_copy_b128_to_lds(unsigned lds_byte_off, const void* g) {
    asm volatile("global_load_async_to_lds_b128 %0, %1, off"
                 :: "v"(lds_byte_off), "v"((unsigned long long)g)
                 : "memory");
}

__device__ inline void wait_async0() {
#if __has_builtin(__builtin_amdgcn_s_wait_asynccnt)
    __builtin_amdgcn_s_wait_asynccnt(0);
#else
    asm volatile("s_wait_asynccnt 0x0" ::: "memory");
#endif
}

__device__ inline float fast_tanh(float x) {
#if __has_builtin(__builtin_amdgcn_tanhf)
    return __builtin_amdgcn_tanhf(x);
#else
    return tanhf(x);
#endif
}

// A 16x32 bf16 fragment (CDNA5 ISA §7.12.2): lane<16 row=lane, K {k0..k0+7, k0+16..k0+23};
// lane>=16 row=lane-16, K base +8.
__device__ inline v16bf load_a_bf16(const __bf16* __restrict__ base, int row,
                                    int ldk, int k0, int lane) {
    const __bf16* p = base + (size_t)row * ldk + k0 + ((lane & 16) ? 8 : 0);
    v8bf lo = *(const v8bf*)(p);
    v8bf hi = *(const v8bf*)(p + 16);
    v16bf r;
#pragma unroll
    for (int i = 0; i < 8; ++i) { r[i] = lo[i]; r[i + 8] = hi[i]; }
    return r;
}

__device__ inline v16bf load_a_f32cvt(const float* __restrict__ base, int row,
                                      int ldk, int k0, int lane) {
    const float* p = base + (size_t)row * ldk + k0 + ((lane & 16) ? 8 : 0);
    v8f lo = *(const v8f*)(p);
    v8f hi = *(const v8f*)(p + 16);
    v16bf r;
#pragma unroll
    for (int i = 0; i < 8; ++i) { r[i] = (__bf16)lo[i]; r[i + 8] = (__bf16)hi[i]; }
    return r;
}

// B 32x16 fragment from LDS-staged, padded column-major-by-column weights:
// lane 0-15: K {k0..k0+7, k0+16..k0+23} of col lane; lane 16-31: K base +8, col lane-16.
__device__ inline v16bf load_b_lds(const __bf16* sW, int ldk, int k0, int lane) {
    const __bf16* p = sW + (size_t)(lane & 15) * ldk + k0 + ((lane & 16) ? 8 : 0);
    v8bf lo = *(const v8bf*)(p);
    v8bf hi = *(const v8bf*)(p + 16);
    v16bf r;
#pragma unroll
    for (int i = 0; i < 8; ++i) { r[i] = lo[i]; r[i + 8] = hi[i]; }
    return r;
}

// ---------------------------------------------------------------------------
// Prep kernels
// ---------------------------------------------------------------------------
__global__ __launch_bounds__(256) void rnn_prep(
    const float* __restrict__ Wi, const float* __restrict__ bi,
    const float* __restrict__ Wh, const float* __restrict__ bh,
    __bf16* __restrict__ Wi_b, __bf16* __restrict__ Wh_b,
    float* __restrict__ bias, __bf16* __restrict__ h0) {
    int i = blockIdx.x * blockDim.x + threadIdx.x;
    if (i < DH * DH)  Wh_b[i] = (__bf16)Wh[i];
    if (i < DH * DIN) Wi_b[i] = (__bf16)Wi[i];
    if (i < BSZ * DH) h0[i]   = (__bf16)0.0f;
    if (i < DH)       bias[i] = bi[i] + bh[i];
}

__global__ __launch_bounds__(256) void cvt_x(const float* __restrict__ x,
                                             __bf16* __restrict__ xb, int n8) {
    int i = blockIdx.x * blockDim.x + threadIdx.x;
    if (i < n8) {
        v8f v = *(const v8f*)(x + (size_t)i * 8);
        v8bf o;
#pragma unroll
        for (int j = 0; j < 8; ++j) o[j] = (__bf16)v[j];
        *(v8bf*)(xb + (size_t)i * 8) = o;
    }
}

// ---------------------------------------------------------------------------
// One step: h_new = tanh(x_t @ Wi^T + h_prev @ Wh^T + bias)
// Grid: 32 blocks (one per 16-wide ntile of DH), 8 waves = 8 mtiles of BSZ.
// Block stages its 16 Wi/Wh columns into LDS with async-to-LDS copies.
// ---------------------------------------------------------------------------
template <bool X_IS_BF16>
__global__ __launch_bounds__(256) void rnn_step(
    const void*   __restrict__ x_t,     // [BSZ, DIN] f32 or bf16
    const __bf16* __restrict__ Wi,      // [DH, DIN] bf16
    const __bf16* __restrict__ Wh,      // [DH, DH]  bf16
    const float*  __restrict__ bias,    // [DH] (bi+bh)
    const __bf16* __restrict__ h_prev,  // [BSZ, DH] bf16
    float*        __restrict__ h_out,   // [BSZ, DH] f32 -> h_seq[t]
    __bf16*       __restrict__ h_next)  // [BSZ, DH] bf16
{
    __shared__ __bf16 sWi[16 * WI_LDK];
    __shared__ __bf16 sWh[16 * WH_LDK];

    const int tid  = threadIdx.x;
    const int lane = tid & 31;
    const int wave = tid >> 5;          // 0..7 == mtile
    const int n0   = blockIdx.x * 16;   // ntile base column
    const int m0   = wave * 16;
    const int arow = m0 + (lane & 15);

    // --- Stage this block's 16 weight columns into LDS (async path) ---
#pragma unroll
    for (int idx = tid; idx < 16 * (DIN / 8); idx += 256) {   // Wi: 512 x 16B
        int col = idx >> 5, kc = idx & 31;
        async_copy_b128_to_lds(
            lds_off_u32(&sWi[col * WI_LDK + kc * 8]),
            Wi + (size_t)(n0 + col) * DIN + kc * 8);
    }
#pragma unroll
    for (int idx = tid; idx < 16 * (DH / 8); idx += 256) {    // Wh: 1024 x 16B
        int col = idx >> 6, kc = idx & 63;
        async_copy_b128_to_lds(
            lds_off_u32(&sWh[col * WH_LDK + kc * 8]),
            Wh + (size_t)(n0 + col) * DH + kc * 8);
    }
    wait_async0();
    __syncthreads();

    v8f acc = {};

    // Input projection: K = DIN
#pragma unroll
    for (int k = 0; k < DIN; k += 32) {
        v16bf a = X_IS_BF16
            ? load_a_bf16((const __bf16*)x_t, arow, DIN, k, lane)
            : load_a_f32cvt((const float*)x_t, arow, DIN, k, lane);
        v16bf b = load_b_lds(sWi, WI_LDK, k, lane);
        acc = __builtin_amdgcn_wmma_f32_16x16x32_bf16(
            false, a, false, b, (short)0, acc, false, false);
    }

    // Recurrence: K = DH
#pragma unroll 4
    for (int k = 0; k < DH; k += 32) {
        v16bf a = load_a_bf16(h_prev, arow, DH, k, lane);
        v16bf b = load_b_lds(sWh, WH_LDK, k, lane);
        acc = __builtin_amdgcn_wmma_f32_16x16x32_bf16(
            false, a, false, b, (short)0, acc, false, false);
    }

    // Epilogue: C/D layout — VGPR r, lanes 0-15 -> M=m0+r, lanes 16-31 -> M=m0+8+r
    const int   n  = n0 + (lane & 15);
    const float bn = bias[n];
    const int   mb = m0 + ((lane >> 4) & 1) * 8;
#pragma unroll
    for (int r = 0; r < 8; ++r) {
        float h = fast_tanh(acc[r] + bn);
        h_out [(size_t)(mb + r) * DH + n] = h;
        h_next[(size_t)(mb + r) * DH + n] = (__bf16)h;
    }
}

__global__ __launch_bounds__(256) void copy_last(
    const float* __restrict__ src, float* __restrict__ dst, int n) {
    int i = blockIdx.x * blockDim.x + threadIdx.x;
    if (i < n) dst[i] = src[i];
}

// ---------------------------------------------------------------------------
extern "C" void kernel_launch(void* const* d_in, const int* in_sizes, int n_in,
                              void* d_out, int out_size, void* d_ws, size_t ws_size,
                              hipStream_t stream) {
    const float* x  = (const float*)d_in[0];  // (SEQ, BSZ, DIN)
    const float* Wi = (const float*)d_in[1];  // (DH, DIN)
    const float* bi = (const float*)d_in[2];  // (DH)
    const float* Wh = (const float*)d_in[3];  // (DH, DH)
    const float* bh = (const float*)d_in[4];  // (DH)
    float* out = (float*)d_out;               // h_seq ++ h_last

    // Workspace layout (bytes)
    char* ws = (char*)d_ws;
    __bf16* Wh_b = (__bf16*)(ws);               // 524288
    __bf16* Wi_b = (__bf16*)(ws + 524288);      // 262144
    __bf16* hA   = (__bf16*)(ws + 786432);      // 131072
    __bf16* hB   = (__bf16*)(ws + 917504);      // 131072
    float*  bias = (float*) (ws + 1048576);     // 2048
    __bf16* x_b  = (__bf16*)(ws + 2097152);     // 33554432
    const size_t WS_NEED_XB = 2097152 + (size_t)SEQ * BSZ * DIN * 2;
    const bool use_xb = ws_size >= WS_NEED_XB;  // deterministic per run

    rnn_prep<<<(DH * DH + 255) / 256, 256, 0, stream>>>(
        Wi, bi, Wh, bh, Wi_b, Wh_b, bias, hA);

    if (use_xb) {
        const int n8 = SEQ * BSZ * DIN / 8;
        cvt_x<<<(n8 + 255) / 256, 256, 0, stream>>>(x, x_b, n8);
    }

    const __bf16* hp = hA;
    __bf16*       hn = hB;
    for (int t = 0; t < SEQ; ++t) {
        float* h_out = out + (size_t)t * BSZ * DH;
        if (use_xb) {
            rnn_step<true><<<32, 256, 0, stream>>>(
                x_b + (size_t)t * BSZ * DIN, Wi_b, Wh_b, bias, hp, h_out, hn);
        } else {
            rnn_step<false><<<32, 256, 0, stream>>>(
                x + (size_t)t * BSZ * DIN, Wi_b, Wh_b, bias, hp, h_out, hn);
        }
        const __bf16* tmp = hn; hn = (__bf16*)hp; hp = tmp;
    }

    copy_last<<<(BSZ * DH + 255) / 256, 256, 0, stream>>>(
        out + (size_t)(SEQ - 1) * BSZ * DH,
        out + (size_t)SEQ * BSZ * DH, BSZ * DH);
}
